// EncoderWDMPNN_87754771792393
// MI455X (gfx1250) — matrix operand order repaired and checked by reading, not compiled, not run on hardware
//
#include <hip/hip_runtime.h>
#include <hip/hip_bf16.h>

typedef __attribute__((ext_vector_type(16))) __bf16 v16bf;
typedef __attribute__((ext_vector_type(8)))  __bf16 v8bf;
typedef __attribute__((ext_vector_type(4)))  __bf16 v4bf;
typedef __attribute__((ext_vector_type(8)))  float  v8f;
typedef __attribute__((ext_vector_type(4)))  unsigned int v4u;
typedef __attribute__((ext_vector_type(8)))  int v8i;
typedef __attribute__((ext_vector_type(4)))  int v4i;

#define HID 512
#define BM 128
#define BN 128
#define BK 32
#define LDA 40   // 32 data + 8 pad bf16 (80 B row stride; TDM pad: 16DW interval + 4DW pad)
#define LDB 40

#define HAVE_TDM __has_builtin(__builtin_amdgcn_tensor_load_to_lds)

// ---------------------------------------------------------------------------
// Cast fp32 weight [512][K] -> bf16 [512][Kp] (zero padded K..Kp)
__global__ void k_cast_pad(const float* __restrict__ W, __bf16* __restrict__ Wb,
                           int K, int Kp, int total) {
    int idx = blockIdx.x * blockDim.x + threadIdx.x;
    if (idx >= total) return;
    int n = idx / Kp;
    int k = idx - n * Kp;
    Wb[idx] = (k < K) ? (__bf16)W[(size_t)n * K + k] : (__bf16)0.0f;
}

// Build A for GEMM1: [V[src[e]] ; E[e] ; 0-pad] as bf16 [nE][160]. block = 160 thr.
__global__ void k_build_Ain(const float* __restrict__ V, const float* __restrict__ Eb,
                            const int* __restrict__ src, __bf16* __restrict__ A) {
    int e = blockIdx.x;
    int k = threadIdx.x;
    float v;
    if (k < 133)       v = V[(size_t)src[e] * 133 + k];
    else if (k < 147)  v = Eb[(size_t)e * 14 + (k - 133)];
    else               v = 0.0f;
    A[(size_t)e * 160 + k] = (__bf16)v;
}

// Build A for GEMM3: [V[n] ; m_final[n] ; 0-pad] as bf16 [nA][672]. block = 672 thr.
__global__ void k_build_A3(const float* __restrict__ V, const float* __restrict__ m2a,
                           __bf16* __restrict__ A) {
    int n = blockIdx.x;
    int k = threadIdx.x;
    float v;
    if (k < 133)       v = V[(size_t)n * 133 + k];
    else if (k < 645)  v = m2a[(size_t)n * HID + (k - 133)];
    else               v = 0.0f;
    A[(size_t)n * 672 + k] = (__bf16)v;
}

// Segment-sum scatter: m2a[dest[e]] += w[e] * h[e]  (f32 atomics, float4 per thread)
__global__ void k_scatter(const float* __restrict__ h, const float* __restrict__ wgt,
                          const int* __restrict__ dest, float* __restrict__ m2a, int nE) {
    int idx = blockIdx.x * blockDim.x + threadIdx.x;
    if (idx >= nE * (HID / 4)) return;
    int e = idx >> 7;
    int k = (idx & 127) << 2;
    float w = wgt ? wgt[e] : 1.0f;
    float4 hv = *(const float4*)(h + (size_t)e * HID + k);
    float* p = m2a + (size_t)dest[e] * HID + k;
    atomicAdd(p + 0, w * hv.x);
    atomicAdd(p + 1, w * hv.y);
    atomicAdd(p + 2, w * hv.z);
    atomicAdd(p + 3, w * hv.w);
}

// Edge message: m[e] = m2a[src[e]] - w[rev[e]] * h[rev[e]]  -> bf16 (GEMM A operand)
__global__ void k_edge_msg(const float* __restrict__ m2a, const float* __restrict__ h,
                           const int* __restrict__ src, const int* __restrict__ rev,
                           const float* __restrict__ wgt, __bf16* __restrict__ Mb, int nE) {
    int idx = blockIdx.x * blockDim.x + threadIdx.x;
    if (idx >= nE * (HID / 4)) return;
    int e = idx >> 7;
    int k = (idx & 127) << 2;
    int r  = rev[e];
    float wr = wgt[r];
    float4 a = *(const float4*)(m2a + (size_t)src[e] * HID + k);
    float4 b = *(const float4*)(h   + (size_t)r      * HID + k);
    v4bf o;
    o.x = (__bf16)(a.x - wr * b.x);
    o.y = (__bf16)(a.y - wr * b.y);
    o.z = (__bf16)(a.z - wr * b.z);
    o.w = (__bf16)(a.w - wr * b.w);
    *(v4bf*)(Mb + (size_t)e * HID + k) = o;
}

__global__ void k_batch(const int* __restrict__ ab, float* __restrict__ out, int n) {
    int i = blockIdx.x * blockDim.x + threadIdx.x;
    if (i < n) out[i] = (float)ab[i];
}

// ---------------------------------------------------------------------------
#if HAVE_TDM
// Issue one TDM 2-D tile load: tile (BK x tileRows) of bf16 from a row-major
// tensor with row stride `strideElems`, into LDS at byte offset `ldsByteOff`,
// with 16 B of LDS padding appended per 64 B row (-> row stride LDA/LDB = 40 bf16).
// Rows beyond `rowsAvail` read as zero (hardware OOB).
__device__ __forceinline__ void tdm_tile_load(const __bf16* gTile, unsigned ldsByteOff,
                                              unsigned rowsAvail, unsigned tileRows,
                                              unsigned strideElems)
{
    unsigned long long ga = (unsigned long long)(const void*)gTile;
    // D# group 0: count=1 | lds_addr | global_addr[56:0] | type=2
    v4u g0 = { 1u,
               ldsByteOff,
               (unsigned)ga,
               (unsigned)((ga >> 32) & 0x1FFFFFFull) | 0x80000000u };
    // D# group 1
    const unsigned dim0  = BK;                 // tensor x extent (tile never exceeds)
    const unsigned dim1  = rowsAvail;          // rows remaining from tile start
    const unsigned tile0 = BK;
    const unsigned tile1 = tileRows;
    // data_size=1 (2B) | pad_enable | pad_interval=3 (16 DW) | pad_amount=3 (4 DW)
    const unsigned d0 = (1u << 16) | (1u << 20) | (3u << 22) | (3u << 25);
    v8i g1 = {
        (int)d0,
        (int)((dim0 & 0xFFFFu) << 16),                                   // dim0[15:0]
        (int)(((dim0 >> 16) & 0xFFFFu) | ((dim1 & 0xFFFFu) << 16)),      // dim0[31:16]|dim1[15:0]
        (int)(((dim1 >> 16) & 0xFFFFu) | ((tile0 & 0xFFFFu) << 16)),     // dim1[31:16]|tile0
        (int)(tile1 & 0xFFFFu),                                          // tile1 | tile2=0
        (int)strideElems,                                                // dim0_stride[31:0]
        0,                                                               // stride hi | dim1_stride lo
        0
    };
    v4i g4z = { 0, 0, 0, 0 };
    v8i g8z = { 0, 0, 0, 0, 0, 0, 0, 0 };
    // 6-arg form on this toolchain: (g0, g1, g2, g3, g_ext, cpol)
    __builtin_amdgcn_tensor_load_to_lds(g0, g1, g4z, g4z, g8z, 0);
}

__device__ __forceinline__ unsigned lds_byte_off(const void* p) {
    // generic pointers to LDS carry the LDS byte offset in their low 32 bits
    return (unsigned)(unsigned long long)p;
}
#endif

// One K-chunk of WMMA math from staged LDS tiles.
__device__ __forceinline__ void wmma_chunk(const __bf16* __restrict__ Ab,
                                           const __bf16* __restrict__ Wb,
                                           int lane, int wv, v8f* acc)
{
    // A fragment (16x32): lanes 0-15 K 0-7 & 16-23; lanes 16-31 K 8-15 & 24-31
    int aRow = wv * 16 + (lane & 15);
    int koA  = (lane & 16) ? 8 : 0;
    const __bf16* ap = Ab + aRow * LDA + koA;
    v8bf alo = *(const v8bf*)(ap);
    v8bf ahi = *(const v8bf*)(ap + 16);
    v16bf afrag = __builtin_shufflevector(alo, ahi,
        0, 1, 2, 3, 4, 5, 6, 7, 8, 9, 10, 11, 12, 13, 14, 15);

    #pragma unroll
    for (int j = 0; j < 8; ++j) {
        int bn  = j * 16 + (lane & 15);
        int koB = (lane & 16) ? 8 : 0;
        const __bf16* bp = Wb + bn * LDB + koB;
        v8bf blo = *(const v8bf*)(bp);
        v8bf bhi = *(const v8bf*)(bp + 16);
        v16bf bfrag = __builtin_shufflevector(blo, bhi,
            0, 1, 2, 3, 4, 5, 6, 7, 8, 9, 10, 11, 12, 13, 14, 15);
        acc[j] = __builtin_amdgcn_wmma_f32_16x16x32_bf16(
            false, afrag, false, bfrag, (short)0, acc[j], false, false);
    }
}

// ---------------------------------------------------------------------------
// Tiled bf16 GEMM + bias + ReLU:  Out[M][512] = relu(A[M][Kp] @ W^T + bias)
// W stored [512][Kp] row-major (already B^T -> lane-contiguous K).
// 256 threads = 8 waves; block tile 128x128, K step 32; wave computes 16x128.
// Staging: TDM double-buffered (wave0 drives DMA, TENSORcnt sync), else sync LDS.
__global__ __launch_bounds__(256) void k_gemm_bf16(
    const __bf16* __restrict__ A, int M, int Kp,
    const __bf16* __restrict__ W,
    const float* __restrict__ biasMat,   // [M][512] or null (h0)
    const float* __restrict__ biasVec,   // [512]    or null (b_o)
    float* __restrict__ Out,             // [M][512]
    float* __restrict__ Out2)            // optional duplicate output or null
{
    const int tid  = threadIdx.x;
    const int lane = tid & 31;
    const int wv   = tid >> 5;
    const int m0   = blockIdx.x * BM;
    const int n0   = blockIdx.y * BN;

    v8f acc[8] = {};

#if HAVE_TDM
    __shared__ __bf16 Alds[2][BM * LDA];
    __shared__ __bf16 Wlds[2][BN * LDB];

    const int nChunks = Kp / BK;
    const unsigned rowsA = (unsigned)(M - m0);
    if (wv == 0) {   // wave-uniform branch: only wave 0 drives the TDM
        tdm_tile_load(A + (size_t)m0 * Kp, lds_byte_off(&Alds[0][0]), rowsA, BM, (unsigned)Kp);
        tdm_tile_load(W + (size_t)n0 * Kp, lds_byte_off(&Wlds[0][0]), BN, BN, (unsigned)Kp);
    }
    for (int i = 0; i < nChunks; ++i) {
        const int buf = i & 1;
        if (wv == 0) {
            if (i + 1 < nChunks) {
                const int kc2 = (i + 1) * BK;
                tdm_tile_load(A + (size_t)m0 * Kp + kc2, lds_byte_off(&Alds[buf ^ 1][0]),
                              rowsA, BM, (unsigned)Kp);
                tdm_tile_load(W + (size_t)n0 * Kp + kc2, lds_byte_off(&Wlds[buf ^ 1][0]),
                              BN, BN, (unsigned)Kp);
                __builtin_amdgcn_s_wait_tensorcnt(2);  // current tile's 2 ops done
            } else {
                __builtin_amdgcn_s_wait_tensorcnt(0);
            }
        }
        __syncthreads();                  // tile `buf` visible to all waves
        wmma_chunk(&Alds[buf][0], &Wlds[buf][0], lane, wv, acc);
        __syncthreads();                  // done reading before it is re-filled
    }
#else
    __shared__ __bf16 Alds[BM * LDA];
    __shared__ __bf16 Wlds[BN * LDB];

    for (int kc = 0; kc < Kp; kc += BK) {
        #pragma unroll
        for (int c = tid; c < BM * 4; c += 256) {
            int row = c >> 2;
            int ko  = (c & 3) * 8;
            int gr  = m0 + row;
            v8bf v = {};
            if (gr < M) v = *(const v8bf*)(A + (size_t)gr * Kp + kc + ko);
            *(v8bf*)(&Alds[row * LDA + ko]) = v;
        }
        #pragma unroll
        for (int c = tid; c < BN * 4; c += 256) {
            int nn = c >> 2;
            int ko = (c & 3) * 8;
            v8bf v = *(const v8bf*)(W + (size_t)(n0 + nn) * Kp + kc + ko);
            *(v8bf*)(&Wlds[nn * LDB + ko]) = v;
        }
        __syncthreads();
        wmma_chunk(Alds, Wlds, lane, wv, acc);
        __syncthreads();
    }
#endif

    // Epilogue: VGPR r: lanes 0-15 -> M=r, lanes 16-31 -> M=r+8; N = lane&15
    int rowBase = m0 + wv * 16 + ((lane & 16) ? 8 : 0);
    int colBase = n0 + (lane & 15);
    #pragma unroll
    for (int j = 0; j < 8; ++j) {
        int col = colBase + j * 16;
        #pragma unroll
        for (int r = 0; r < 8; ++r) {
            int row = rowBase + r;
            if (row < M) {
                float v = acc[j][r];
                if (biasMat) v += biasMat[(size_t)row * HID + col];
                if (biasVec) v += biasVec[col];
                v = fmaxf(v, 0.0f);
                Out[(size_t)row * HID + col] = v;
                if (Out2) Out2[(size_t)row * HID + col] = v;
            }
        }
    }
}

// ---------------------------------------------------------------------------
extern "C" void kernel_launch(void* const* d_in, const int* in_sizes, int n_in,
                              void* d_out, int out_size, void* d_ws, size_t ws_size,
                              hipStream_t stream) {
    const float* V   = (const float*)d_in[0];
    const float* Eb  = (const float*)d_in[1];
    const int*   ei  = (const int*)d_in[2];
    const int*   rev = (const int*)d_in[3];
    const int*   ab  = (const int*)d_in[4];
    const float* wgt = (const float*)d_in[5];
    const float* W_i = (const float*)d_in[6];
    const float* W_h = (const float*)d_in[7];
    const float* W_o = (const float*)d_in[8];
    const float* b_o = (const float*)d_in[9];

    const int nA = in_sizes[0] / 133;   // atoms
    const int nE = in_sizes[1] / 14;    // directed edges
    const int* src = ei;
    const int* dst = ei + nE;

    // ---- workspace carve-out ----
    char* ws = (char*)d_ws;
    size_t off = 0;
    auto carve = [&](size_t bytes) -> void* {
        void* p = ws + off;
        off = (off + bytes + 255) & ~(size_t)255;
        return p;
    };
    __bf16* Wi_b = (__bf16*)carve((size_t)512 * 160 * 2);
    __bf16* Wh_b = (__bf16*)carve((size_t)512 * 512 * 2);
    __bf16* Wo_b = (__bf16*)carve((size_t)512 * 672 * 2);
    float*  h0   = (float*)carve((size_t)nE * HID * 4);
    float*  hbuf = (float*)carve((size_t)nE * HID * 4);
    float*  m2a  = (float*)carve((size_t)nA * HID * 4);
    // shared region, reused sequentially: Ain -> m_bf16 -> A3
    size_t shmax = (size_t)nE * HID * 2;
    __bf16* Ain = (__bf16*)carve(shmax);
    __bf16* Mb  = Ain;
    __bf16* A3  = Ain;

    float* dout = (float*)d_out;
    float* hfin = dout + (size_t)nA * HID + nA;          // final h goes straight to d_out

    // ---- weight casts ----
    k_cast_pad<<<(512 * 160 + 255) / 256, 256, 0, stream>>>(W_i, Wi_b, 147, 160, 512 * 160);
    k_cast_pad<<<(512 * 512 + 255) / 256, 256, 0, stream>>>(W_h, Wh_b, 512, 512, 512 * 512);
    k_cast_pad<<<(512 * 672 + 255) / 256, 256, 0, stream>>>(W_o, Wo_b, 645, 672, 512 * 672);

    // ---- GEMM1: h0 = relu([V[src];E] @ W_i^T); also copy into hbuf ----
    k_build_Ain<<<nE, 160, 0, stream>>>(V, Eb, src, Ain);
    dim3 gE((nE + BM - 1) / BM, HID / BN);
    k_gemm_bf16<<<gE, 256, 0, stream>>>(Ain, nE, 160, Wi_b, nullptr, nullptr, h0, hbuf);

    // ---- T message-passing steps ----
    const int scatGrid = (nE * (HID / 4) + 255) / 256;
    for (int t = 0; t < 3; ++t) {
        (void)hipMemsetAsync(m2a, 0, (size_t)nA * HID * 4, stream);
        k_scatter<<<scatGrid, 256, 0, stream>>>(hbuf, wgt, dst, m2a, nE);
        k_edge_msg<<<scatGrid, 256, 0, stream>>>(m2a, hbuf, src, rev, wgt, Mb, nE);
        float* out = (t == 2) ? hfin : hbuf;
        k_gemm_bf16<<<gE, 256, 0, stream>>>(Mb, nE, 512, Wh_b, h0, nullptr, out, nullptr);
    }

    // ---- readout: m_final = segment_sum(h, dest); h_atom = relu([V;m_final] @ W_o^T + b_o)
    (void)hipMemsetAsync(m2a, 0, (size_t)nA * HID * 4, stream);
    k_scatter<<<scatGrid, 256, 0, stream>>>(hfin, nullptr, dst, m2a, nE);
    k_build_A3<<<nA, 672, 0, stream>>>(V, m2a, A3);
    dim3 gA((nA + BM - 1) / BM, HID / BN);
    k_gemm_bf16<<<gA, 256, 0, stream>>>(A3, nA, 672, Wo_b, nullptr, b_o, dout, nullptr);

    // ---- atom_batch passthrough (exact in f32: values < 256) ----
    k_batch<<<(nA + 255) / 256, 256, 0, stream>>>(ab, dout + (size_t)nA * HID, nA);
}
